// HMAELayer_87514253623565
// MI455X (gfx1250) — matrix operand order repaired and checked by reading, not compiled, not run on hardware
//
#include <hip/hip_runtime.h>

#define NN 2048
#define PP 3
#define RR 5
#define EPSV 1e-6f

typedef __attribute__((ext_vector_type(16))) _Float16 v16h;
typedef __attribute__((ext_vector_type(8)))  _Float16 v8h;
typedef __attribute__((ext_vector_type(8)))  float    v8f;

union V16U { v16h v; v8h h[2]; };
union V16E { v8h h[2]; _Float16 e[16]; };

// ---------------------------------------------------------------------------
// Kernel 1: per-(p,v) column sums of masked exp(S) for both weight sets.
// Thread = one column v; loops 128 rows u; 5 coalesced adj reads per row.
// ---------------------------------------------------------------------------
__global__ __launch_bounds__(256) void colsum_kernel(const float* __restrict__ adj,
                                                     const float* __restrict__ w1,
                                                     const float* __restrict__ w2,
                                                     float* __restrict__ denomA,
                                                     float* __restrict__ denomB) {
  const int v  = blockIdx.x * 256 + threadIdx.x;
  const int u0 = blockIdx.y * 128;

  float W1r[PP * RR], W2r[PP * RR];
#pragma unroll
  for (int i = 0; i < PP * RR; ++i) { W1r[i] = w1[i]; W2r[i] = w2[i]; }

  float sA[PP] = {0.f, 0.f, 0.f};
  float sB[PP] = {0.f, 0.f, 0.f};

  for (int du = 0; du < 128; ++du) {
    const int u = u0 + du;
    float av[RR];
    bool mask = false;
#pragma unroll
    for (int r = 0; r < RR; ++r) {
      av[r] = adj[((size_t)r * NN + u) * NN + v];
      mask = mask || (av[r] != 0.0f);
    }
    if (mask) {
#pragma unroll
      for (int p = 0; p < PP; ++p) {
        float s1 = 0.f, s2 = 0.f;
#pragma unroll
        for (int r = 0; r < RR; ++r) {
          s1 = fmaf(W1r[p * RR + r], av[r], s1);
          s2 = fmaf(W2r[p * RR + r], av[r], s2);
        }
        sA[p] += __expf(s1);
        sB[p] += __expf(s2);
      }
    }
  }
#pragma unroll
  for (int p = 0; p < PP; ++p) {
    atomicAdd(&denomA[p * NN + v], sA[p]);
    atomicAdd(&denomB[p * NN + v], sB[p]);
  }
}

// ---------------------------------------------------------------------------
// Kernel 2: recompute E, normalize. Writes norm_a f32 (output), norm_a/norm_b
// f16 (GEMM inputs in ws). Also copies W1/W2 to the output tuple slots.
// ---------------------------------------------------------------------------
__global__ __launch_bounds__(256) void normalize_kernel(const float* __restrict__ adj,
                                                        const float* __restrict__ w1,
                                                        const float* __restrict__ w2,
                                                        const float* __restrict__ denomA,
                                                        const float* __restrict__ denomB,
                                                        float* __restrict__ outNormA,
                                                        _Float16* __restrict__ Ah,
                                                        _Float16* __restrict__ Bh,
                                                        float* __restrict__ outW1,
                                                        float* __restrict__ outW2) {
  const int v  = blockIdx.x * 256 + threadIdx.x;
  const int u0 = blockIdx.y * 128;

  if (blockIdx.x == 0 && blockIdx.y == 0 && threadIdx.x < PP * RR) {
    outW1[threadIdx.x] = w1[threadIdx.x];
    outW2[threadIdx.x] = w2[threadIdx.x];
  }

  float W1r[PP * RR], W2r[PP * RR];
#pragma unroll
  for (int i = 0; i < PP * RR; ++i) { W1r[i] = w1[i]; W2r[i] = w2[i]; }

  float invA[PP], invB[PP];
#pragma unroll
  for (int p = 0; p < PP; ++p) {
    invA[p] = 1.0f / (denomA[p * NN + v] + EPSV);
    invB[p] = 1.0f / (denomB[p * NN + v] + EPSV);
  }

  for (int du = 0; du < 128; ++du) {
    const int u = u0 + du;
    float av[RR];
    bool mask = false;
#pragma unroll
    for (int r = 0; r < RR; ++r) {
      av[r] = adj[((size_t)r * NN + u) * NN + v];
      mask = mask || (av[r] != 0.0f);
    }
#pragma unroll
    for (int p = 0; p < PP; ++p) {
      float na = 0.f, nb = 0.f;
      if (mask) {
        float s1 = 0.f, s2 = 0.f;
#pragma unroll
        for (int r = 0; r < RR; ++r) {
          s1 = fmaf(W1r[p * RR + r], av[r], s1);
          s2 = fmaf(W2r[p * RR + r], av[r], s2);
        }
        na = __expf(s1) * invA[p];
        nb = __expf(s2) * invB[p];
      }
      const size_t idx = ((size_t)p * NN + u) * NN + v;
      outNormA[idx] = na;
      Ah[idx] = (_Float16)na;
      Bh[idx] = (_Float16)nb;
    }
  }
}

// ---------------------------------------------------------------------------
// Kernel 3: prod[p] = norm_a[p] (f16) @ norm_b[p] (f16), f32 accumulate.
// Block tile 128x128, K step 32, 8 waves (wave32), each wave does 32x64 via
// 2x4 v_wmma_f32_16x16x32_f16 accumulators fed from LDS.
// ---------------------------------------------------------------------------
#define LDT 40  // padded LDS row stride in halves (80B, 16B-aligned)

__global__ __launch_bounds__(256) void gemm_kernel(const _Float16* __restrict__ Ahg,
                                                   const _Float16* __restrict__ Bhg,
                                                   float* __restrict__ Cg) {
  __shared__ __align__(16) _Float16 lA[128][LDT];
  __shared__ __align__(16) _Float16 lB[128][LDT];  // stored transposed: [w][k]

  const int p  = blockIdx.z;
  const _Float16* A = Ahg + (size_t)p * NN * NN;
  const _Float16* B = Bhg + (size_t)p * NN * NN;
  float* C = Cg + (size_t)p * NN * NN;

  const int m0 = blockIdx.y * 128;
  const int n0 = blockIdx.x * 128;

  const int tid  = threadIdx.x;
  const int lane = tid & 31;
  const int wave = tid >> 5;
  const int wm = wave & 3;   // 4 waves along M -> 32 rows each
  const int wn = wave >> 2;  // 2 waves along N -> 64 cols each
  const int lm = lane & 15;
  const int hs = lane >> 4;

  v8f acc[2][4];
  const v8f vzero = {0.f, 0.f, 0.f, 0.f, 0.f, 0.f, 0.f, 0.f};
#pragma unroll
  for (int m = 0; m < 2; ++m)
#pragma unroll
    for (int n = 0; n < 4; ++n) acc[m][n] = vzero;

  // global->LDS mapping (256 threads, 32B per thread per tile)
  const int arow = tid >> 1, acol = (tid & 1) * 16;
  const int bk = tid >> 3, bw = (tid & 7) * 16;

  for (int k0 = 0; k0 < NN; k0 += 32) {
    // A tile: rows m0..m0+127, cols k0..k0+31 (row-major in LDS)
    {
      const v8h* src = (const v8h*)(A + (size_t)(m0 + arow) * NN + k0 + acol);
      *(v8h*)&lA[arow][acol]     = src[0];
      *(v8h*)&lA[arow][acol + 8] = src[1];
    }
    // B tile: rows k0..k0+31, cols n0..n0+127; coalesced read, transpose in LDS
    {
      const v8h* src = (const v8h*)(B + (size_t)(k0 + bk) * NN + n0 + bw);
      V16E t;
      t.h[0] = src[0];
      t.h[1] = src[1];
#pragma unroll
      for (int i = 0; i < 16; ++i) lB[bw + i][bk] = t.e[i];
    }
    // prefetch next K tiles (gfx1250 global_prefetch_b8)
    if (k0 + 32 < NN) {
      __builtin_prefetch(A + (size_t)(m0 + arow) * NN + k0 + 32 + acol, 0, 0);
      __builtin_prefetch(B + (size_t)(k0 + 32 + bk) * NN + n0 + bw, 0, 0);
    }
    __syncthreads();

    // A frag (ISA 16-bit A layout): lane<16 K{0..7,16..23}, lane>=16 K{8..15,24..31}
    V16U af[2];
#pragma unroll
    for (int m = 0; m < 2; ++m) {
      const int r = wm * 32 + m * 16 + lm;
      af[m].h[0] = *(const v8h*)&lA[r][hs * 8];
      af[m].h[1] = *(const v8h*)&lA[r][16 + hs * 8];
    }
    // B frag: lane<16 col=lm K{0..7,16..23}, lane>=16 col=lm K{8..15,24..31}
    V16U bf[4];
#pragma unroll
    for (int n = 0; n < 4; ++n) {
      const int c = wn * 64 + n * 16 + lm;
      bf[n].h[0] = *(const v8h*)&lB[c][hs * 8];
      bf[n].h[1] = *(const v8h*)&lB[c][16 + hs * 8];
    }
#pragma unroll
    for (int m = 0; m < 2; ++m)
#pragma unroll
      for (int n = 0; n < 4; ++n)
        acc[m][n] = __builtin_amdgcn_wmma_f32_16x16x32_f16(
            false, af[m].v, false, bf[n].v, (short)0, acc[m][n], false, false);

    __syncthreads();
  }

  // Epilogue: C VGPR j -> row (+8 for lanes>=16), col = lm
#pragma unroll
  for (int m = 0; m < 2; ++m) {
    const int rbase = m0 + wm * 32 + m * 16 + hs * 8;
#pragma unroll
    for (int n = 0; n < 4; ++n) {
      const int ccol = n0 + wn * 64 + n * 16 + lm;
#pragma unroll
      for (int j = 0; j < 8; ++j)
        C[(size_t)(rbase + j) * NN + ccol] = acc[m][n][j];
    }
  }
}

// ---------------------------------------------------------------------------
extern "C" void kernel_launch(void* const* d_in, const int* in_sizes, int n_in,
                              void* d_out, int out_size, void* d_ws, size_t ws_size,
                              hipStream_t stream) {
  (void)in_sizes; (void)n_in; (void)out_size; (void)ws_size;
  const float* adj = (const float*)d_in[0];
  const float* w1  = (const float*)d_in[1];
  const float* w2  = (const float*)d_in[2];
  float* out = (float*)d_out;

  // output tuple layout (flattened, return order): prod, W1, W2, norm_a
  const size_t prod_elems = (size_t)PP * NN * NN;  // 12,582,912
  float* out_prod  = out;
  float* out_W1    = out + prod_elems;
  float* out_W2    = out + prod_elems + PP * RR;
  float* out_normA = out + prod_elems + 2 * (PP * RR);

  // workspace layout
  float* denomA = (float*)d_ws;                                  // 3*2048 f32
  float* denomB = denomA + (size_t)PP * NN;                      // 3*2048 f32
  _Float16* Ah  = (_Float16*)((char*)d_ws + 65536);              // 24 MB f16
  _Float16* Bh  = Ah + prod_elems;                               // 24 MB f16

  hipMemsetAsync(d_ws, 0, (size_t)2 * PP * NN * sizeof(float), stream);

  dim3 gridE(NN / 256, NN / 128);
  colsum_kernel<<<gridE, 256, 0, stream>>>(adj, w1, w2, denomA, denomB);
  normalize_kernel<<<gridE, 256, 0, stream>>>(adj, w1, w2, denomA, denomB,
                                              out_normA, Ah, Bh, out_W1, out_W2);

  dim3 gridG(NN / 128, NN / 128, PP);
  gemm_kernel<<<gridG, 256, 0, stream>>>(Ah, Bh, out_prod);
}
